// Attention_63359357551381
// MI455X (gfx1250) — matrix operand order repaired and checked by reading, not compiled, not run on hardware
//
#include <hip/hip_runtime.h>

typedef __attribute__((ext_vector_type(16))) __bf16 v16bf;
typedef __attribute__((ext_vector_type(8)))  __bf16 v8bf;
typedef __attribute__((ext_vector_type(4)))  __bf16 v4bf;
typedef __attribute__((ext_vector_type(8)))  float  v8f;

namespace {
constexpr int Sseq  = 2048;
constexpr int Dh    = 128;
constexpr int NH    = 32;
constexpr int KVH   = 8;
constexpr int RATIO = NH / KVH;      // 4
constexpr int BM    = 64;            // queries per block (4 waves x 16)
constexpr int BN    = 64;            // keys per tile iteration
constexpr int KSTR  = Dh + 8;        // 136 bf16 (272 B) row stride -> conflict-free
constexpr int VSTR  = BN + 8;        // 72 bf16 (144 B)
constexpr int PSTR  = BN + 8;        // 72 bf16
constexpr float NEGV = -1.0e6f;

union U16bf { v16bf v; v8bf h[2]; };

__device__ __forceinline__ v16bf combine16(v8bf lo, v8bf hi) {
  U16bf u; u.h[0] = lo; u.h[1] = hi; return u.v;
}
} // namespace

__global__ __launch_bounds__(128)
void attn_prefill_wmma(const float* __restrict__ qg,
                       const float* __restrict__ kg,
                       const float* __restrict__ vg,
                       const float* __restrict__ sinks,
                       const int*   __restrict__ bwp,
                       float* __restrict__ outg)
{
  __shared__ __align__(16) __bf16 Klds[BN * KSTR];       // 17408 B, row-major K tile
  __shared__ __align__(16) __bf16 Vlds[Dh * VSTR];       // 18432 B, transposed V tile (d-major)
  __shared__ __align__(16) __bf16 Plds[4 * 16 * PSTR];   //  9216 B, per-wave P scratch

  const int tid  = threadIdx.x;
  const int wave = tid >> 5;
  const int lane = tid & 31;
  const int l16  = lane & 15;
  const bool hi  = (lane >= 16);

  const int q0  = blockIdx.x * BM;
  const int h   = blockIdx.y;
  const int kvh = h / RATIO;
  const int bw  = bwp[0];

  const float scale = 0.08838834764831845f; // 1/sqrt(128), folded into Q

  // ---------------- Q A-fragments, kept in VGPRs for the whole kernel -------
  // A layout (16-bit, 16x32): lane(0-15)=row M, elems 0..7 -> K=base..base+7,
  // elems 8..15 -> K=base+16..base+23, base = chunk*32 + (hi?8:0).
  const int qrow = q0 + wave * 16 + l16;
  const float* qp = qg + ((size_t)h * Sseq + qrow) * Dh;
  v16bf qfrag[4];
#pragma unroll
  for (int c = 0; c < 4; ++c) {
    const int dbase = c * 32 + (hi ? 8 : 0);
    U16bf u;
#pragma unroll
    for (int g = 0; g < 2; ++g) {
      const float4 a = *(const float4*)(qp + dbase + g * 16);
      const float4 b = *(const float4*)(qp + dbase + g * 16 + 4);
      v8bf t;
      t[0] = (__bf16)(a.x * scale); t[1] = (__bf16)(a.y * scale);
      t[2] = (__bf16)(a.z * scale); t[3] = (__bf16)(a.w * scale);
      t[4] = (__bf16)(b.x * scale); t[5] = (__bf16)(b.y * scale);
      t[6] = (__bf16)(b.z * scale); t[7] = (__bf16)(b.w * scale);
      u.h[g] = t;
    }
    qfrag[c] = u.v;
  }

  // ---------------- online-softmax state (sink-initialized) -----------------
  const float sink = sinks[h];
  float m_i[8], l_i[8];
  v8f o[8];
  v8f vzero = {};
#pragma unroll
  for (int r = 0; r < 8; ++r) { m_i[r] = sink; l_i[r] = 1.0f; o[r] = vzero; }

  int kt_lo = 0;
  if (bw > 0) { const int lo = q0 - bw + 1; if (lo > 0) kt_lo = lo / BN; }
  const int kt_hi = (q0 + BM - 1) / BN;

  const float* kbasep = kg + (size_t)kvh * Sseq * Dh;
  const float* vbasep = vg + (size_t)kvh * Sseq * Dh;

  for (int kt = kt_lo; kt <= kt_hi; ++kt) {
    const int kb = kt * BN;
    __syncthreads();
    // -------- stage K (row-major) and V (transposed) tiles as bf16 ----------
    const float* kp = kbasep + (size_t)kb * Dh;
    const float* vp = vbasep + (size_t)kb * Dh;
#pragma unroll
    for (int i = 0; i < 16; ++i) {
      const int idx4 = i * 128 + tid;          // 2048 float4 per 64x128 tile
      const int row  = idx4 >> 5;              // 32 float4 per row
      const int col  = (idx4 & 31) * 4;
      const float4 kk = *(const float4*)(kp + row * Dh + col);
      v4bf kq;
      kq[0] = (__bf16)kk.x; kq[1] = (__bf16)kk.y;
      kq[2] = (__bf16)kk.z; kq[3] = (__bf16)kk.w;
      *(v4bf*)&Klds[row * KSTR + col] = kq;
      const float4 vv = *(const float4*)(vp + row * Dh + col);
      Vlds[(col + 0) * VSTR + row] = (__bf16)vv.x;
      Vlds[(col + 1) * VSTR + row] = (__bf16)vv.y;
      Vlds[(col + 2) * VSTR + row] = (__bf16)vv.z;
      Vlds[(col + 3) * VSTR + row] = (__bf16)vv.w;
    }
    __syncthreads();

    // -------- S = Q K^T : 16x64 per wave (4 N-tiles x 4 d-chunks) -----------
    // B layout (16-bit, 32x16): lane -> key N=l16, elems 0..7 -> d = dbase..dbase+7,
    // elems 8..15 -> d = dbase+16..dbase+23, dbase = chunk*32 + (hi?8:0).
    v8f s[4];
#pragma unroll
    for (int n = 0; n < 4; ++n) {
      v8f acc = {};
#pragma unroll
      for (int c = 0; c < 4; ++c) {
        const __bf16* kr = &Klds[(n * 16 + l16) * KSTR + c * 32 + (hi ? 8 : 0)];
        v16bf bmat = combine16(*(const v8bf*)kr, *(const v8bf*)(kr + 16));
        acc = __builtin_amdgcn_wmma_f32_16x16x32_bf16(
            false, qfrag[c], false, bmat, (short)0, acc, false, false);
      }
      s[n] = acc;
    }

    // -------- causal + sliding-window mask at C-layout coords ---------------
    // C layout: lane -> N=l16 (+16 group), VGPR r -> M = r + (hi?8:0).
#pragma unroll
    for (int n = 0; n < 4; ++n) {
      const int kpos = kb + n * 16 + l16;
#pragma unroll
      for (int r = 0; r < 8; ++r) {
        const int qpos = q0 + wave * 16 + r + (hi ? 8 : 0);
        const bool msk = (kpos > qpos) || (bw > 0 && kpos < qpos - bw + 1);
        if (msk) s[n][r] = NEGV;
      }
    }

    // -------- online softmax (rows live in 16-lane half-waves) --------------
    float rmax[8];
#pragma unroll
    for (int r = 0; r < 8; ++r)
      rmax[r] = fmaxf(fmaxf(s[0][r], s[1][r]), fmaxf(s[2][r], s[3][r]));
#pragma unroll
    for (int xm = 1; xm < 16; xm <<= 1)
#pragma unroll
      for (int r = 0; r < 8; ++r)
        rmax[r] = fmaxf(rmax[r], __shfl_xor(rmax[r], xm, 32));

#pragma unroll
    for (int r = 0; r < 8; ++r) {
      const float mn    = fmaxf(m_i[r], rmax[r]);
      const float alpha = __expf(m_i[r] - mn);
      m_i[r] = mn;
      l_i[r] *= alpha;
#pragma unroll
      for (int dt = 0; dt < 8; ++dt) o[dt][r] *= alpha;
    }
    float rsum[8];
#pragma unroll
    for (int r = 0; r < 8; ++r) rsum[r] = 0.0f;
#pragma unroll
    for (int n = 0; n < 4; ++n)
#pragma unroll
      for (int r = 0; r < 8; ++r) {
        const float p = __expf(s[n][r] - m_i[r]);
        s[n][r] = p;
        rsum[r] += p;
      }
#pragma unroll
    for (int xm = 1; xm < 16; xm <<= 1)
#pragma unroll
      for (int r = 0; r < 8; ++r)
        rsum[r] += __shfl_xor(rsum[r], xm, 32);
#pragma unroll
    for (int r = 0; r < 8; ++r) l_i[r] += rsum[r];

    // -------- P: C-layout regs -> per-wave LDS -> A-layout fragments --------
    __bf16* Pl = &Plds[wave * 16 * PSTR];
#pragma unroll
    for (int n = 0; n < 4; ++n)
#pragma unroll
      for (int r = 0; r < 8; ++r) {
        const int M = r + (hi ? 8 : 0);
        Pl[M * PSTR + n * 16 + l16] = (__bf16)s[n][r];
      }
    v16bf pf[2];
#pragma unroll
    for (int ch = 0; ch < 2; ++ch) {
      const __bf16* pr = &Pl[l16 * PSTR + ch * 32 + (hi ? 8 : 0)];
      pf[ch] = combine16(*(const v8bf*)pr, *(const v8bf*)(pr + 16));
    }

    // -------- O += P V : 8 d-tiles x 2 key-chunks ---------------------------
    // B fragment from transposed V: elems 0..7 -> key kbase+j, elems 8..15 -> key kbase+16+(j-8).
#pragma unroll
    for (int dt = 0; dt < 8; ++dt) {
      const int d = dt * 16 + l16;
#pragma unroll
      for (int ch = 0; ch < 2; ++ch) {
        const __bf16* vr = &Vlds[d * VSTR + ch * 32 + (hi ? 8 : 0)];
        v16bf bmat = combine16(*(const v8bf*)vr, *(const v8bf*)(vr + 16));
        o[dt] = __builtin_amdgcn_wmma_f32_16x16x32_bf16(
            false, pf[ch], false, bmat, (short)0, o[dt], false, false);
      }
    }
  }

  // ---------------- epilogue: out = O / l ------------------------------------
#pragma unroll
  for (int r = 0; r < 8; ++r) {
    const float inv  = 1.0f / l_i[r];
    const int qpos   = q0 + wave * 16 + r + (hi ? 8 : 0);
    float* op = outg + ((size_t)h * Sseq + qpos) * Dh;
#pragma unroll
    for (int dt = 0; dt < 8; ++dt)
      op[dt * 16 + l16] = o[dt][r] * inv;
  }
}

extern "C" void kernel_launch(void* const* d_in, const int* in_sizes, int n_in,
                              void* d_out, int out_size, void* d_ws, size_t ws_size,
                              hipStream_t stream) {
  (void)in_sizes; (void)n_in; (void)out_size; (void)d_ws; (void)ws_size;
  const float* q     = (const float*)d_in[0];
  const float* k     = (const float*)d_in[1];
  const float* v     = (const float*)d_in[2];
  const float* sinks = (const float*)d_in[3];
  const int*   bw    = (const int*)d_in[4];
  float* out = (float*)d_out;

  dim3 grid(Sseq / BM, NH, 1);   // (32, 32)
  dim3 block(128, 1, 1);         // 4 waves of 32
  attn_prefill_wmma<<<grid, block, 0, stream>>>(q, k, v, sinks, bw, out);
}
